// LSTMClassifierQuantized_80934363726416
// MI455X (gfx1250) — 8-bit matrix operand order repaired and checked by reading, compiled, not run on hardware
//
#include <hip/hip_runtime.h>
#include <hip/hip_bf16.h>
#include <math.h>


typedef int v8i __attribute__((ext_vector_type(8)));

static constexpr int BB   = 64;
static constexpr int SEQ  = 512;
static constexpr int EMB  = 512;
static constexpr int HID  = 512;
static constexpr int G4   = 2048;
static constexpr int KT   = 8;
static constexpr int NWG  = 16;
static constexpr int TPB  = 256;

__device__ __forceinline__ v8i wmma_iu8(v8i a, v8i b, v8i c) {
  return __builtin_amdgcn_wmma_i32_16x16x64_iu8(true, a, true, b, c, false, false);
}

__device__ __forceinline__ int q8(float x, float inv_s) {
  int q = (int)rintf(x * inv_s);
  return q < -127 ? -127 : (q > 127 ? 127 : q);
}

__device__ __forceinline__ void grid_sync(unsigned* bar, unsigned* gen, unsigned nwg) {
  __syncthreads();
  if (threadIdx.x == 0) {
    __threadfence();
    unsigned g = __hip_atomic_load(gen, __ATOMIC_ACQUIRE, __HIP_MEMORY_SCOPE_AGENT);
    unsigned prev = __hip_atomic_fetch_add(bar, 1u, __ATOMIC_ACQ_REL, __HIP_MEMORY_SCOPE_AGENT);
    if (prev == nwg - 1u) {
      __hip_atomic_store(bar, 0u, __ATOMIC_RELAXED, __HIP_MEMORY_SCOPE_AGENT);
      __hip_atomic_fetch_add(gen, 1u, __ATOMIC_RELEASE, __HIP_MEMORY_SCOPE_AGENT);
    } else {
      while (__hip_atomic_load(gen, __ATOMIC_ACQUIRE, __HIP_MEMORY_SCOPE_AGENT) == g) {
        __builtin_amdgcn_s_sleep(2);
      }
    }
  }
  __syncthreads();
}

__global__ void init_kernel(float* hbuf, unsigned* hmaxb, unsigned* cmaxb,
                            unsigned* wmaxb, unsigned* barp, int* lenb,
                            const int* __restrict__ mask) {
  int idx = blockIdx.x * TPB + threadIdx.x;
  if (idx < BB * HID) hbuf[idx] = 0.0f;
  if (idx < SEQ + 1) { hmaxb[idx] = 0u; cmaxb[idx] = 0u; }
  if (idx < 2) { wmaxb[idx] = 0u; barp[idx] = 0u; }
  if (idx < BB) {
    int s = 0;
    for (int t = 0; t < SEQ; ++t) s += mask[idx * SEQ + t];
    lenb[idx] = s - 1;
  }
}

__global__ void wmax_kernel(const float* __restrict__ W, int n, unsigned* out) {
  int tid = blockIdx.x * TPB + threadIdx.x;
  float m = 0.0f;
  for (int i = tid; i < n; i += gridDim.x * TPB) m = fmaxf(m, fabsf(W[i]));
  for (int o = 16; o; o >>= 1) m = fmaxf(m, __shfl_xor(m, o, 32));
  if ((threadIdx.x & 31) == 0) atomicMax(out, __float_as_uint(m));
}

__global__ void pack_w_kernel(const float* __restrict__ W,
                              const unsigned* __restrict__ maxb, int msel,
                              unsigned* __restrict__ out) {
  int idx = blockIdx.x * TPB + threadIdx.x;
  int tile = idx >> 8;
  int tn = tile >> 3, tk = tile & 7;
  int l = (idx >> 3) & 31, v = idx & 7;
  int n  = tn * 16 + (l & 15);
  int kb = tk * 64 + (v >> 1) * 16 + (l < 16 ? 0 : 8) + (v & 1) * 4;
  float s = fmaxf(__uint_as_float(maxb[msel]), 1e-8f) / 127.0f;
  float inv = 1.0f / s;
  unsigned d = 0;
  for (int bt = 0; bt < 4; ++bt) {
    int q = q8(W[(size_t)n * 512 + kb + bt], inv);
    d |= (unsigned)(q & 0xFF) << (8 * bt);
  }
  out[idx] = d;
}

__global__ void xscale_kernel(const int* __restrict__ ids,
                              const float* __restrict__ emb,
                              float* __restrict__ sx) {
  int t = blockIdx.x;
  int tid = threadIdx.x;
  float m = 0.0f;
  for (int i = tid; i < BB * EMB; i += TPB) {
    int b = i >> 9, e = i & 511;
    m = fmaxf(m, fabsf(emb[(size_t)ids[b * SEQ + t] * EMB + e]));
  }
  __shared__ float red[8];
  for (int o = 16; o; o >>= 1) m = fmaxf(m, __shfl_xor(m, o, 32));
  if ((tid & 31) == 0) red[tid >> 5] = m;
  __syncthreads();
  if (tid == 0) {
    float mm = red[0];
    for (int i = 1; i < 8; ++i) mm = fmaxf(mm, red[i]);
    sx[t] = fmaxf(mm, 1e-8f) / 127.0f;
  }
}

__global__ void pack_x_kernel(const int* __restrict__ ids,
                              const float* __restrict__ emb,
                              const float* __restrict__ sx,
                              unsigned* __restrict__ Xq) {
  int idx = blockIdx.x * TPB + threadIdx.x;
  int tile = idx >> 8;
  int tm = tile >> 3, tk = tile & 7;
  int l = (idx >> 3) & 31, v = idx & 7;
  int row = tm * 16 + (l & 15);
  int t = row >> 6, b = row & 63;
  int kb = tk * 64 + (v >> 1) * 16 + (l < 16 ? 0 : 8) + (v & 1) * 4;
  float inv = 1.0f / sx[t];
  const float* erow = emb + (size_t)ids[b * SEQ + t] * EMB;
  unsigned d = 0;
  for (int bt = 0; bt < 4; ++bt) {
    int q = q8(erow[kb + bt], inv);
    d |= (unsigned)(q & 0xFF) << (8 * bt);
  }
  Xq[idx] = d;
}

__global__ void gemm_x_kernel(const unsigned char* __restrict__ Xq,
                              const unsigned char* __restrict__ BiP,
                              const float* __restrict__ sx,
                              const unsigned* __restrict__ wmaxb,
                              const float* __restrict__ b_inp,
                              const float* __restrict__ b_hid,
                              float* __restrict__ gates_x) {
  int lane = threadIdx.x & 31;
  int wave = threadIdx.x >> 5;
  int job = blockIdx.x * 8 + wave;
  int tm2 = job >> 6;
  int tn2 = job & 63;
  float swi = fmaxf(__uint_as_float(wmaxb[0]), 1e-8f) / 127.0f;

  const unsigned char* ap0 = Xq  + (size_t)(tm2 * 2 + 0) * KT * 1024 + lane * 32;
  const unsigned char* ap1 = Xq  + (size_t)(tm2 * 2 + 1) * KT * 1024 + lane * 32;
  const unsigned char* bp0 = BiP + (size_t)(tn2 * 2 + 0) * KT * 1024 + lane * 32;
  const unsigned char* bp1 = BiP + (size_t)(tn2 * 2 + 1) * KT * 1024 + lane * 32;

  v8i acc00 = {}, acc01 = {}, acc10 = {}, acc11 = {};
#pragma unroll
  for (int tk = 0; tk < KT; ++tk) {
    v8i a0 = *(const v8i*)(ap0 + (size_t)tk * 1024);
    v8i a1 = *(const v8i*)(ap1 + (size_t)tk * 1024);
    v8i b0 = *(const v8i*)(bp0 + (size_t)tk * 1024);
    v8i b1 = *(const v8i*)(bp1 + (size_t)tk * 1024);
    acc00 = wmma_iu8(a0, b0, acc00);
    acc01 = wmma_iu8(a0, b1, acc01);
    acc10 = wmma_iu8(a1, b0, acc10);
    acc11 = wmma_iu8(a1, b1, acc11);
  }
  int t = tm2 >> 1;
  float scale = sx[t] * swi;
#pragma unroll
  for (int mi = 0; mi < 2; ++mi) {
#pragma unroll
    for (int ni = 0; ni < 2; ++ni) {
      v8i acc = (mi == 0) ? (ni == 0 ? acc00 : acc01)
                          : (ni == 0 ? acc10 : acc11);
      int col = (tn2 * 2 + ni) * 16 + (lane & 15);
      float bias = b_inp[col] + b_hid[col];
#pragma unroll
      for (int i = 0; i < 8; ++i) {
        int row = (tm2 * 2 + mi) * 16 + i + (lane < 16 ? 0 : 8);
        gates_x[(size_t)row * G4 + col] = (float)acc[i] * scale + bias;
      }
    }
  }
}

__global__ void lstm_recurrent_kernel(const float* __restrict__ gates_x,
                                      const unsigned char* __restrict__ BhP,
                                      const unsigned* __restrict__ wmaxb,
                                      float* __restrict__ hbuf,
                                      float* __restrict__ hlast,
                                      unsigned* hmaxb, unsigned* cmaxb,
                                      const int* __restrict__ lenb,
                                      unsigned* barp) {
  extern __shared__ char smem[];
  char*  sWh = smem;
  char*  sA  = smem + (64 << 10);
  float* sg  = (float*)(smem + (96 << 10));

  const int tid  = threadIdx.x;
  const int lane = tid & 31;
  const int wave = tid >> 5;
  const int wg   = blockIdx.x;

  const float s_wh = fmaxf(__uint_as_float(wmaxb[1]), 1e-8f) / 127.0f;

  for (int i = tid; i < 4096; i += TPB) {
    int nt = i >> 9;
    int rem = i & 511;
    int g = nt >> 1, p = nt & 1;
    int gtn = g * 32 + wg * 2 + p;
    unsigned lds_off = (unsigned)(uintptr_t)sWh + (unsigned)i * 16u;
    unsigned long long gaddr =
        (unsigned long long)(uintptr_t)(BhP + (size_t)gtn * KT * 1024 + (size_t)rem * 16);
    asm volatile("global_load_async_to_lds_b128 %0, %1, off"
                 :: "v"(lds_off), "v"(gaddr) : "memory");
  }
  asm volatile("s_wait_asynccnt 0x0" ::: "memory");

  float creg[8];
#pragma unroll
  for (int k = 0; k < 8; ++k) creg[k] = 0.0f;
  __syncthreads();

  for (int t = 0; t < SEQ; ++t) {
    {
      int pb = tid >> 2, pg = tid & 3;
      __builtin_prefetch(&gates_x[((size_t)t * BB + pb) * G4 + pg * 512 + wg * 32], 0, 1);
    }

    float ssh = fmaxf(__uint_as_float(hmaxb[t]), 1e-8f) / 127.0f;
    float inv_sh = 1.0f / ssh;
    for (int i = tid; i < 8192; i += TPB) {
      int tile = i >> 8;
      int tm = tile >> 3, tk = tile & 7;
      int l = (i >> 3) & 31, v = i & 7;
      int row = tm * 16 + (l & 15);
      int kb = tk * 64 + (v >> 1) * 16 + (l < 16 ? 0 : 8) + (v & 1) * 4;
      unsigned d = 0;
      for (int bt = 0; bt < 4; ++bt) {
        int q = q8(hbuf[row * HID + kb + bt], inv_sh);
        d |= (unsigned)(q & 0xFF) << (8 * bt);
      }
      ((unsigned*)sA)[i] = d;
    }
    __syncthreads();

    float ssc = fmaxf(__uint_as_float(cmaxb[t]), 1e-8f) / 127.0f;

#pragma unroll
    for (int j = 0; j < 4; ++j) {
      int idx = wave * 4 + j;
      int tm = idx & 3, nt = idx >> 2;
      v8i acc = {};
#pragma unroll
      for (int tk = 0; tk < KT; ++tk) {
        v8i av = *(const v8i*)(sA  + ((tm * KT + tk) << 10) + lane * 32);
        v8i bv = *(const v8i*)(sWh + ((nt * KT + tk) << 10) + lane * 32);
        acc = wmma_iu8(av, bv, acc);
      }
      int g = nt >> 1, p = nt & 1;
      int lcol = g * 32 + p * 16 + (lane & 15);
      int gcol = g * 512 + wg * 32 + p * 16 + (lane & 15);
      float dq = ssh * s_wh;
#pragma unroll
      for (int i = 0; i < 8; ++i) {
        int bb = tm * 16 + i + (lane < 16 ? 0 : 8);
        sg[bb * 128 + lcol] = (float)acc[i] * dq
                            + gates_x[((size_t)t * BB + bb) * G4 + gcol];
      }
    }
    __syncthreads();

    float mh = 0.0f, mc = 0.0f;
#pragma unroll
    for (int k = 0; k < 8; ++k) {
      int e = tid + k * TPB;
      int bb = e >> 5, lc = e & 31;
      float ig = sg[bb * 128 +       lc];
      float fg = sg[bb * 128 +  32 + lc];
      float gg = sg[bb * 128 +  64 + lc];
      float og = sg[bb * 128 +  96 + lc];
      ig = 1.0f / (1.0f + expf(-ig));
      fg = 1.0f / (1.0f + expf(-fg));
      gg = tanhf(gg);
      og = 1.0f / (1.0f + expf(-og));
      float cq = rintf(creg[k] / ssc) * ssc;
      float cn = fg * cq + ig * gg;
      float hn = og * tanhf(cn);
      creg[k] = cn;
      hbuf[bb * HID + wg * 32 + lc] = hn;
      if (t == lenb[bb]) hlast[bb * HID + wg * 32 + lc] = hn;
      mh = fmaxf(mh, fabsf(hn));
      mc = fmaxf(mc, fabsf(cn));
    }
    for (int o = 16; o; o >>= 1) {
      mh = fmaxf(mh, __shfl_xor(mh, o, 32));
      mc = fmaxf(mc, __shfl_xor(mc, o, 32));
    }
    if (lane == 0) {
      atomicMax(&hmaxb[t + 1], __float_as_uint(mh));
      atomicMax(&cmaxb[t + 1], __float_as_uint(mc));
    }
    grid_sync(&barp[0], &barp[1], NWG);
  }
}

__global__ void final_kernel(const float* __restrict__ hlast,
                             const float* __restrict__ lin_w,
                             const float* __restrict__ lin_b,
                             float* __restrict__ out) {
  __shared__ float red[256];
  __shared__ float sv[2];
  int tid = threadIdx.x;

  float m = 0.0f;
  for (int i = tid; i < BB * HID; i += TPB) m = fmaxf(m, fabsf(hlast[i]));
  red[tid] = m; __syncthreads();
  for (int o = 128; o; o >>= 1) { if (tid < o) red[tid] = fmaxf(red[tid], red[tid + o]); __syncthreads(); }
  if (tid == 0) sv[0] = fmaxf(red[0], 1e-8f) / 127.0f;
  __syncthreads();

  m = 0.0f;
  for (int i = tid; i < HID; i += TPB) m = fmaxf(m, fabsf(lin_w[i]));
  red[tid] = m; __syncthreads();
  for (int o = 128; o; o >>= 1) { if (tid < o) red[tid] = fmaxf(red[tid], red[tid + o]); __syncthreads(); }
  if (tid == 0) sv[1] = fmaxf(red[0], 1e-8f) / 127.0f;
  __syncthreads();

  float sh = sv[0], sw = sv[1];
  int b = tid >> 2, part = tid & 3;
  float acc = 0.0f;
  for (int j = part * 128; j < part * 128 + 128; ++j) {
    float qh = rintf(hlast[b * HID + j] / sh) * sh;
    float qw = rintf(lin_w[j] / sw) * sw;
    acc += qh * qw;
  }
  red[tid] = acc; __syncthreads();
  if (part == 0)
    out[b] = red[tid] + red[tid + 1] + red[tid + 2] + red[tid + 3] + lin_b[0];
}

extern "C" void kernel_launch(void* const* d_in, const int* in_sizes, int n_in,
                              void* d_out, int out_size, void* d_ws, size_t ws_size,
                              hipStream_t stream) {
  (void)in_sizes; (void)n_in; (void)out_size; (void)ws_size;
  const int*   ids   = (const int*)d_in[0];
  const int*   mask  = (const int*)d_in[1];
  const float* emb   = (const float*)d_in[2];
  const float* W_inp = (const float*)d_in[3];
  const float* W_hid = (const float*)d_in[4];
  const float* b_inp = (const float*)d_in[5];
  const float* b_hid = (const float*)d_in[6];
  const float* lin_w = (const float*)d_in[7];
  const float* lin_b = (const float*)d_in[8];
  float* out = (float*)d_out;

  char* w = (char*)d_ws;
  auto take = [&](size_t bytes) {
    char* p = w; w += (bytes + 255) & ~(size_t)255; return p;
  };
  float*         gates_x = (float*)take((size_t)SEQ * BB * G4 * sizeof(float));
  unsigned char* Xq      = (unsigned char*)take((size_t)SEQ * BB * EMB);
  unsigned char* BiP     = (unsigned char*)take((size_t)G4 * EMB);
  unsigned char* BhP     = (unsigned char*)take((size_t)G4 * HID);
  float*         sx      = (float*)take(SEQ * sizeof(float));
  unsigned*      wmaxb   = (unsigned*)take(2 * sizeof(unsigned));
  float*         hbuf    = (float*)take(BB * HID * sizeof(float));
  float*         hlast   = (float*)take(BB * HID * sizeof(float));
  unsigned*      hmaxb   = (unsigned*)take((SEQ + 1) * sizeof(unsigned));
  unsigned*      cmaxb   = (unsigned*)take((SEQ + 1) * sizeof(unsigned));
  int*           lenb    = (int*)take(BB * sizeof(int));
  unsigned*      barp    = (unsigned*)take(2 * sizeof(unsigned));

  hipFuncSetAttribute((const void*)lstm_recurrent_kernel,
                      hipFuncAttributeMaxDynamicSharedMemorySize, 128 << 10);

  init_kernel<<<128, TPB, 0, stream>>>(hbuf, hmaxb, cmaxb, wmaxb, barp, lenb, mask);

  wmax_kernel<<<256, TPB, 0, stream>>>(W_inp, G4 * EMB, &wmaxb[0]);
  wmax_kernel<<<256, TPB, 0, stream>>>(W_hid, G4 * HID, &wmaxb[1]);
  pack_w_kernel<<<1024, TPB, 0, stream>>>(W_inp, wmaxb, 0, (unsigned*)BiP);
  pack_w_kernel<<<1024, TPB, 0, stream>>>(W_hid, wmaxb, 1, (unsigned*)BhP);

  xscale_kernel<<<SEQ, TPB, 0, stream>>>(ids, emb, sx);
  pack_x_kernel<<<16384, TPB, 0, stream>>>(ids, emb, sx, (unsigned*)Xq);

  gemm_x_kernel<<<8192, TPB, 0, stream>>>(Xq, BiP, sx, wmaxb, b_inp, b_hid, gates_x);

  lstm_recurrent_kernel<<<NWG, TPB, 128 << 10, stream>>>(
      gates_x, BhP, wmaxb, hbuf, hlast, hmaxb, cmaxb, lenb, barp);

  final_kernel<<<1, TPB, 0, stream>>>(hlast, lin_w, lin_b, out);
}
